// HierClusterVectorQuantizerEMA_3281355014170
// MI455X (gfx1250) — matrix operand order repaired and checked by reading, not compiled, not run on hardware
//
#include <hip/hip_runtime.h>
#include <hip/hip_bf16.h>

typedef __attribute__((ext_vector_type(16))) _Float16 v16h;
typedef __attribute__((ext_vector_type(8)))  _Float16 v8h;
typedef __attribute__((ext_vector_type(8)))  float    v8f;

#define NE    8192
#define CD    64
#define BB    256
#define HWQ   64
#define NELEM 1048576   // B*C*H*W = 256*64*8*8

// ---------------------------------------------------------------------------
// Embedding prep: fp32 -> f16 copy + squared norms (done once)
// ---------------------------------------------------------------------------
__global__ void k_emb_prep(const float* __restrict__ emb, _Float16* __restrict__ emb16,
                           float* __restrict__ e2) {
  int j = blockIdx.x * blockDim.x + threadIdx.x;
  if (j >= NE) return;
  const float* r = emb + (size_t)j * CD;
  _Float16* o = emb16 + (size_t)j * CD;
  float s = 0.f;
#pragma unroll
  for (int c = 0; c < CD; ++c) { float v = r[c]; s += v * v; o[c] = (_Float16)v; }
  e2[j] = s;
}

// ---------------------------------------------------------------------------
// NCHW fp32 residual -> row-major (B*HW, C) f16 queries
// ---------------------------------------------------------------------------
__global__ void k_cvt_zflat(const float* __restrict__ zr, _Float16* __restrict__ q16) {
  int t = blockIdx.x * blockDim.x + threadIdx.x;
  if (t >= NELEM) return;
  int c = t & 63, pos = (t >> 6) & 63, b = t >> 12;
  q16[t] = (_Float16)zr[(size_t)b * 4096 + c * 64 + pos];  // t = (b*64+pos)*64 + c
}

// ---------------------------------------------------------------------------
// Global mean pool (pn==1 scale): (B,C) means -> f16 queries
// ---------------------------------------------------------------------------
__global__ void k_meanpool(const float* __restrict__ zr, _Float16* __restrict__ zd16) {
  int t = blockIdx.x * blockDim.x + threadIdx.x;  // b*64 + c
  if (t >= BB * CD) return;
  int c = t & 63, b = t >> 6;
  const float* p = zr + (size_t)b * 4096 + c * 64;
  float s = 0.f;
#pragma unroll
  for (int i = 0; i < 64; ++i) s += p[i];
  zd16[t] = (_Float16)(s * (1.f / 64.f));
}

// ---------------------------------------------------------------------------
// Fused WMMA distance GEMM + argmin over all 8192 embeddings.
// One wave = 16 query rows; streams embedding tiles of 16 columns.
// score = z.e - 0.5*|e|^2  (argmax score == argmin d2; |z|^2 cancels)
// ---------------------------------------------------------------------------
__device__ inline v16h loadA(const _Float16* rowp, int kb, int hf) {
  // A layout: lanes 0-15 hold K = kb+{0..15},
  // lanes 16-31 hold K = kb+{16..31}.
  return *(const v16h*)(rowp + kb + hf * 16);
}

__global__ __launch_bounds__(256) void k_nearest(
    const _Float16* __restrict__ q16, const _Float16* __restrict__ emb16,
    const float* __restrict__ e2, int M, int* __restrict__ oidx) {
  const int lane = threadIdx.x & 31;
  const int r = lane & 15, hf = lane >> 4;
  const int row0 = (blockIdx.x * 8 + (threadIdx.x >> 5)) * 16;
  if (row0 >= M) return;  // wave-uniform: EXEC stays all-ones for WMMA

  const _Float16* qrow = q16 + (size_t)(row0 + r) * CD;
  v16h a0 = loadA(qrow, 0, hf);    // K = 0..31
  v16h a1 = loadA(qrow, 32, hf);   // K = 32..63

  float best[8];
  int   bidx[8];
#pragma unroll
  for (int v = 0; v < 8; ++v) { best[v] = -3.4e38f; bidx[v] = 0; }

  for (int j0 = 0; j0 < NE; j0 += 16) {
    // B layout: lane n holds column j0+n; lanes 0-15 K-low half, 16-31 K-high.
    const _Float16* erow = emb16 + (size_t)(j0 + r) * CD + (hf << 4);
    __builtin_prefetch((const void*)(erow + 8 * CD), 0, 1);
    v16h b0 = *(const v16h*)(erow);        // K slice [0,32)
    v16h b1 = *(const v16h*)(erow + 32);   // K slice [32,64)
    float h2 = 0.5f * e2[j0 + r];

    v8f acc = {};
    acc = __builtin_amdgcn_wmma_f32_16x16x32_f16(false, a0, false, b0,
                                                 (short)0, acc, false, false);
    acc = __builtin_amdgcn_wmma_f32_16x16x32_f16(false, a1, false, b1,
                                                 (short)0, acc, false, false);
    int col = j0 + r;
#pragma unroll
    for (int v = 0; v < 8; ++v) {
      float s = acc[v] - h2;                 // lane holds (row v+8*hf, col r)
      if (s > best[v]) { best[v] = s; bidx[v] = col; }
    }
  }

  // Cross-lane reduce over the 16 column slots within each half-wave.
#pragma unroll
  for (int v = 0; v < 8; ++v) {
    float m = best[v]; int bi = bidx[v];
#pragma unroll
    for (int st = 8; st >= 1; st >>= 1) {
      float om = __shfl_xor(m, st, 32);
      int   oi = __shfl_xor(bi, st, 32);
      if (om > m || (om == m && oi < bi)) { m = om; bi = oi; }
    }
    if (r == 0) oidx[row0 + hf * 8 + v] = bi;
  }
}

// ---------------------------------------------------------------------------
// Histogram scatter
// ---------------------------------------------------------------------------
__global__ void k_scatter(const int* __restrict__ idx, int n, float* __restrict__ counts) {
  int t = blockIdx.x * blockDim.x + threadIdx.x;
  if (t < n) atomicAdd(counts + idx[t], 1.f);
}

// ---------------------------------------------------------------------------
// top-k over 8192 counts (single block, selection via LDS argmax passes)
// ---------------------------------------------------------------------------
__global__ __launch_bounds__(256) void k_topk(const float* __restrict__ counts,
                                              int pn, int* __restrict__ topv) {
  __shared__ float cbuf[NE];
  __shared__ float rv[256];
  __shared__ int   ri[256];
  int t = threadIdx.x;
  for (int j = t; j < NE; j += 256) cbuf[j] = counts[j];
  __syncthreads();
  for (int it = 0; it < pn; ++it) {
    float bv = -3.4e38f; int bi = NE;
    for (int j = t; j < NE; j += 256) {
      float v = cbuf[j];
      if (v > bv || (v == bv && j < bi)) { bv = v; bi = j; }
    }
    rv[t] = bv; ri[t] = bi; __syncthreads();
    for (int s = 128; s >= 1; s >>= 1) {
      if (t < s) {
        if (rv[t + s] > rv[t] || (rv[t + s] == rv[t] && ri[t + s] < ri[t])) {
          rv[t] = rv[t + s]; ri[t] = ri[t + s];
        }
      }
      __syncthreads();
    }
    if (t == 0) { topv[it] = ri[0]; cbuf[ri[0]] = -3.4e38f; }
    __syncthreads();
  }
}

// ---------------------------------------------------------------------------
// Per-row softmax over pn selected centroids (one block per row)
// ---------------------------------------------------------------------------
__global__ __launch_bounds__(64) void k_probs(
    const float* __restrict__ zr, const float* __restrict__ emb,
    const float* __restrict__ e2, const int* __restrict__ topv,
    int pn, float* __restrict__ probs) {
  __shared__ float zi[64];
  __shared__ float sv[64];
  int i = blockIdx.x;          // row: b*64 + pos
  int t = threadIdx.x;
  int b = i >> 6, pos = i & 63;
  zi[t] = zr[(size_t)b * 4096 + t * 64 + pos];
  __syncthreads();
  float s = -3.4e38f;
  if (t < pn) {
    int j = topv[t];
    const float* er = emb + (size_t)j * CD;
    float dot = 0.f, z2 = 0.f;
#pragma unroll
    for (int c = 0; c < CD; ++c) { dot += zi[c] * er[c]; z2 += zi[c] * zi[c]; }
    s = -(z2 + e2[j] - 2.f * dot);  // TEMP = 1
  }
  sv[t] = s; __syncthreads();
  for (int st = 32; st >= 1; st >>= 1) { if (t < st && sv[t + st] > sv[t]) sv[t] = sv[t + st]; __syncthreads(); }
  float mx = sv[0]; __syncthreads();
  float e = (t < pn) ? __expf(s - mx) : 0.f;
  sv[t] = e; __syncthreads();
  for (int st = 32; st >= 1; st >>= 1) { if (t < st) sv[t] += sv[t + st]; __syncthreads(); }
  float inv = 1.f / sv[0];
  if (t < pn) probs[(size_t)i * pn + t] = e * inv;
}

__global__ void k_denom(const float* __restrict__ probs, int pn, float* __restrict__ denom) {
  int t = blockIdx.x * blockDim.x + threadIdx.x;  // b*pn + p
  if (t >= BB * pn) return;
  int b = t / pn, p = t % pn;
  const float* pp = probs + (size_t)b * HWQ * pn + p;
  float s = 0.f;
  for (int pos = 0; pos < HWQ; ++pos) s += pp[pos * pn];
  denom[t] = fmaxf(s, 1e-5f);
}

__global__ void k_zdown(const float* __restrict__ probs, const float* __restrict__ zr,
                        const float* __restrict__ denom, int pn, _Float16* __restrict__ zd16) {
  int t = blockIdx.x * blockDim.x + threadIdx.x;  // (b*pn+p)*64 + c
  if (t >= BB * pn * CD) return;
  int c = t & 63; int bp = t >> 6; int b = bp / pn, p = bp % pn;
  const float* pp = probs + (size_t)b * HWQ * pn + p;
  const float* zz = zr + (size_t)b * 4096 + c * 64;
  float s = 0.f;
  for (int pos = 0; pos < HWQ; ++pos) s += pp[pos * pn] * zz[pos];
  zd16[t] = (_Float16)(s / denom[bp]);
}

__global__ void k_recon_mid(const float* __restrict__ probs, const int* __restrict__ idx,
                            const float* __restrict__ emb, int pn, float* __restrict__ zori) {
  int t = blockIdx.x * blockDim.x + threadIdx.x;  // (b*64+pos)*64 + c
  if (t >= NELEM) return;
  int c = t & 63, pos = (t >> 6) & 63, b = t >> 12;
  const float* pp = probs + ((size_t)(b * 64 + pos)) * pn;
  float s = 0.f;
  for (int p = 0; p < pn; ++p) s += pp[p] * emb[(size_t)idx[b * pn + p] * CD + c];
  zori[(size_t)b * 4096 + c * 64 + pos] = s;
}

__global__ void k_bcast1(const int* __restrict__ idx, const float* __restrict__ emb,
                         float* __restrict__ zori) {
  int t = blockIdx.x * blockDim.x + threadIdx.x;  // NCHW flat
  if (t >= NELEM) return;
  int c = (t >> 6) & 63, b = t >> 12;
  zori[t] = emb[(size_t)idx[b] * CD + c];
}

__global__ void k_recon64(const int* __restrict__ idx, const float* __restrict__ emb,
                          float* __restrict__ zori) {
  int t = blockIdx.x * blockDim.x + threadIdx.x;  // NCHW flat
  if (t >= NELEM) return;
  int pos = t & 63, c = (t >> 6) & 63, b = t >> 12;
  zori[t] = emb[(size_t)idx[b * 64 + pos] * CD + c];
}

// ---------------------------------------------------------------------------
// Fused 3x3 conv + alpha-mix + residual updates (z_hat += res; z_rest -= res)
// ---------------------------------------------------------------------------
__global__ __launch_bounds__(256) void k_conv(
    const float* __restrict__ zori, const float* __restrict__ cw,
    const float* __restrict__ cb, float* __restrict__ zhat, float* __restrict__ zrest) {
  int t = blockIdx.x * blockDim.x + threadIdx.x;
  if (t >= NELEM) return;
  int wq = t & 7, hq = (t >> 3) & 7, o = (t >> 6) & 63, b = t >> 12;
  const float* xb = zori + (size_t)b * 4096;
  const float* wb = cw + (size_t)o * 576;
  float y = cb[o];
  for (int i = 0; i < 64; ++i) {
    const float* xi = xb + i * 64;
    const float* wi = wb + i * 9;
#pragma unroll
    for (int kh = 0; kh < 3; ++kh) {
      int hh = hq + kh - 1;
      if ((unsigned)hh >= 8u) continue;
#pragma unroll
      for (int kw = 0; kw < 3; ++kw) {
        int ww = wq + kw - 1;
        if ((unsigned)ww >= 8u) continue;
        y += xi[hh * 8 + ww] * wi[kh * 3 + kw];
      }
    }
  }
  float res = 0.5f * zori[t] + 0.5f * y;   // ALPHA = 0.5
  zhat[t] += res;
  zrest[t] -= res;
}

__global__ __launch_bounds__(256) void k_loss(const float* __restrict__ zh,
                                              const float* __restrict__ z,
                                              float* __restrict__ acc) {
  __shared__ float s[256];
  int t = threadIdx.x;
  float p = 0.f;
  for (size_t i = (size_t)blockIdx.x * blockDim.x + t; i < NELEM;
       i += (size_t)gridDim.x * blockDim.x) {
    float d = zh[i] - z[i];
    p += d * d;
  }
  s[t] = p; __syncthreads();
  for (int st = 128; st >= 1; st >>= 1) { if (t < st) s[t] += s[t + st]; __syncthreads(); }
  if (t == 0) atomicAdd(acc, 0.25f * s[0] * (1.f / (float)NELEM));  // BETA*mean
}

__global__ void k_final(const float* __restrict__ zhat, const float* __restrict__ lossacc,
                        const float* __restrict__ tcnt, float* __restrict__ out) {
  int t = blockIdx.x * blockDim.x + threadIdx.x;
  if (t < NELEM) out[t] = zhat[t];                          // z_hat (straight-through == value)
  else if (t == NELEM) out[t] = lossacc[0] * (1.f / 8.f);   // mean_vq_loss
  else if (t < NELEM + 1 + NE) out[t] = tcnt[t - NELEM - 1];
}

// ---------------------------------------------------------------------------
extern "C" void kernel_launch(void* const* d_in, const int* in_sizes, int n_in,
                              void* d_out, int out_size, void* d_ws, size_t ws_size,
                              hipStream_t stream) {
  (void)in_sizes; (void)n_in; (void)out_size; (void)ws_size;
  const float* z   = (const float*)d_in[0];
  const float* emb = (const float*)d_in[1];
  const float* cw  = (const float*)d_in[2];
  const float* cb  = (const float*)d_in[3];
  float* out = (float*)d_out;

  char* base = (char*)d_ws; size_t off = 0;
  auto walloc = [&](size_t bytes) -> void* {
    void* p = base + off; off = (off + bytes + 255) & ~(size_t)255; return p;
  };
  _Float16* emb16 = (_Float16*)walloc((size_t)NE * CD * 2);
  float*    e2    = (float*)walloc((size_t)NE * 4);
  float*    zrest = (float*)walloc((size_t)NELEM * 4);
  float*    zhat  = (float*)walloc((size_t)NELEM * 4);
  _Float16* q16   = (_Float16*)walloc((size_t)16384 * CD * 2);
  _Float16* zd16  = (_Float16*)walloc((size_t)16384 * CD * 2);
  int*      idx   = (int*)walloc((size_t)16384 * 4);
  float*    cfull = (float*)walloc((size_t)NE * 4);
  float*    tcnt  = (float*)walloc((size_t)NE * 4);
  int*      topv  = (int*)walloc(256);
  float*    probs = (float*)walloc((size_t)16384 * 64 * 4);
  float*    denom = (float*)walloc((size_t)16384 * 4);
  float*    zori  = (float*)walloc((size_t)NELEM * 4);
  float*    lossacc = (float*)walloc(256);

  hipMemcpyAsync(zrest, z, (size_t)NELEM * 4, hipMemcpyDeviceToDevice, stream);
  hipMemsetAsync(zhat, 0, (size_t)NELEM * 4, stream);
  hipMemsetAsync(tcnt, 0, (size_t)NE * 4, stream);
  hipMemsetAsync(lossacc, 0, 4, stream);

  k_emb_prep<<<(NE + 255) / 256, 256, 0, stream>>>(emb, emb16, e2);

  static const int PNS[8] = {1, 4, 9, 16, 25, 36, 49, 64};
  for (int si = 0; si < 8; ++si) {
    int pn = PNS[si];
    if (pn == 1) {
      k_meanpool<<<(BB * CD + 255) / 256, 256, 0, stream>>>(zrest, zd16);
      k_nearest<<<(BB + 127) / 128, 256, 0, stream>>>(zd16, emb16, e2, BB, idx);
      k_scatter<<<(BB + 255) / 256, 256, 0, stream>>>(idx, BB, tcnt);
      k_bcast1<<<(NELEM + 255) / 256, 256, 0, stream>>>(idx, emb, zori);
    } else if (pn == 64) {
      k_cvt_zflat<<<(NELEM + 255) / 256, 256, 0, stream>>>(zrest, q16);
      k_nearest<<<16384 / 128, 256, 0, stream>>>(q16, emb16, e2, 16384, idx);
      k_scatter<<<(16384 + 255) / 256, 256, 0, stream>>>(idx, 16384, tcnt);
      k_recon64<<<(NELEM + 255) / 256, 256, 0, stream>>>(idx, emb, zori);
    } else {
      k_cvt_zflat<<<(NELEM + 255) / 256, 256, 0, stream>>>(zrest, q16);
      k_nearest<<<16384 / 128, 256, 0, stream>>>(q16, emb16, e2, 16384, idx);
      hipMemsetAsync(cfull, 0, (size_t)NE * 4, stream);
      k_scatter<<<(16384 + 255) / 256, 256, 0, stream>>>(idx, 16384, cfull);
      k_topk<<<1, 256, 0, stream>>>(cfull, pn, topv);
      k_probs<<<16384, 64, 0, stream>>>(zrest, emb, e2, topv, pn, probs);
      k_denom<<<(BB * pn + 255) / 256, 256, 0, stream>>>(probs, pn, denom);
      k_zdown<<<(BB * pn * CD + 255) / 256, 256, 0, stream>>>(probs, zrest, denom, pn, zd16);
      int M = BB * pn;
      k_nearest<<<(M + 127) / 128, 256, 0, stream>>>(zd16, emb16, e2, M, idx);
      k_scatter<<<(M + 255) / 256, 256, 0, stream>>>(idx, M, tcnt);
      k_recon_mid<<<(NELEM + 255) / 256, 256, 0, stream>>>(probs, idx, emb, pn, zori);
    }
    k_conv<<<(NELEM + 255) / 256, 256, 0, stream>>>(
        zori, cw + (size_t)si * 64 * 64 * 9, cb + si * 64, zhat, zrest);
    k_loss<<<256, 256, 0, stream>>>(zhat, z, lossacc);
  }
  k_final<<<(NELEM + 1 + NE + 255) / 256, 256, 0, stream>>>(zhat, lossacc, tcnt, out);
}
